// LipformerCharacterDecoder_21947282883224
// MI455X (gfx1250) — matrix operand order repaired and checked by reading, not compiled, not run on hardware
//
#include <hip/hip_runtime.h>

// ---------------------------------------------------------------------------
// LipformerCharacterDecoder for MI455X (gfx1250, wave32, WMMA 16x16x32 bf16)
//
// Plan:
//  - Convert inputs + all weights to bf16; weights stored TRANSPOSED [N][K]
//    so a B-fragment (32x16 K-major per ISA layout) is 2x b128 loads/lane.
//  - MX = inputs @ W_v + b_in precomputed as one big WMMA GEMM (parallel in T).
//  - Encoder: 100 launches; 16 WGs, each owns a 16-row stripe of h:
//    MH = h @ U_v (WMMA, A staged bf16 in LDS), gating in-workgroup.
//  - Decoder: per step, kernel 1 = softmax(pred)@emb -> x@W_p -> h@U_p ->
//    gating (all LDS-local); kernel 2 = q@state^T, softmax, P@state,
//    context, [state|context]@W3 + softmax -> d_out.
// ---------------------------------------------------------------------------

typedef __attribute__((ext_vector_type(16))) __bf16 v16bf;
typedef __attribute__((ext_vector_type(8)))  float  v8f;

#define B_   256
#define T_   100
#define D_   512
#define U_   256
#define V_   64
#define TH3  768   // 3*U

union FragU { uint4 u[2]; v16bf v; };

// A fragment: 16x32 bf16, A row-major in LDS with leading dim lda (halves).
// Lane L<16: row M=L, K = k0 + {0..7, 16..23}; lane>=16: K = k0 + {8..15, 24..31}.
__device__ __forceinline__ v16bf load_a(const __bf16* A, int lda, int k0, int lane) {
    int m  = lane & 15;
    int kb = (lane < 16) ? 0 : 8;
    const __bf16* p = A + m * lda + k0 + kb;
    FragU f;
    f.u[0] = *(const uint4*)(p);        // K = k0+kb .. +7
    f.u[1] = *(const uint4*)(p + 16);   // K = k0+kb+16 .. +23
    return f.v;
}

// B fragment: 32x16 bf16 from a TRANSPOSED matrix BT[N][K] in global memory.
// Lane: N = n0 + (lane&15); lane<16: K = k0 + {0..7, 16..23}; lane>=16: K = k0 + {8..15, 24..31}.
__device__ __forceinline__ v16bf load_bt(const __bf16* BT, int K, int n0, int k0, int lane) {
    int n  = n0 + (lane & 15);
    int kb = (lane >> 4) << 3;          // 0 or 8
    const __bf16* p = BT + (size_t)n * K + k0 + kb;
    FragU f;
    f.u[0] = *(const uint4*)(p);
    f.u[1] = *(const uint4*)(p + 16);
    return f.v;
}

__device__ __forceinline__ v8f wmma_bf16(v16bf a, v16bf b, v8f c) {
    return __builtin_amdgcn_wmma_f32_16x16x32_bf16(
        /*neg_a=*/false, a, /*neg_b=*/false, b,
        /*c_mod=*/(short)0, c, /*reuse_a=*/false, /*reuse_b=*/false);
}

// One wave computes a 16x16 f32 tile of  A[16xK] @ B[KxN] + bias(col).
__device__ __forceinline__ v8f gemm_rowtile(const __bf16* A, int lda,
                                            const __bf16* BT, int K,
                                            int n0, float bias, int lane) {
    v8f acc;
#pragma unroll
    for (int r = 0; r < 8; ++r) acc[r] = bias;
#pragma unroll 4
    for (int k0 = 0; k0 < K; k0 += 32) {
        v16bf a = load_a(A, lda, k0, lane);
        v16bf b = load_bt(BT, K, n0, k0, lane);
        acc = wmma_bf16(a, b, acc);
    }
    return acc;
}

// C/D layout: lane<16 -> (M=r, N=lane); lane>=16 -> (M=r+8, N=lane-16).
__device__ __forceinline__ void store_c_lds(float* C, int ldc, int n0, v8f acc, int lane) {
    int n  = n0 + (lane & 15);
    int mb = (lane < 16) ? 0 : 8;
#pragma unroll
    for (int r = 0; r < 8; ++r) C[(mb + r) * ldc + n] = acc[r];
}

__device__ __forceinline__ float sigm(float x) { return 1.0f / (1.0f + __expf(-x)); }

// ------------------------------- converts ----------------------------------

__global__ void cvt_bf16_kernel(const float* __restrict__ in, __bf16* __restrict__ out, int n) {
    int i = blockIdx.x * blockDim.x + threadIdx.x;
    if (i < n) out[i] = (__bf16)in[i];
}

// in: [K][N] row-major f32  ->  out: [N][K] bf16 (transposed)
__global__ void cvtT_bf16_kernel(const float* __restrict__ in, __bf16* __restrict__ out,
                                 int K, int N) {
    int i = blockIdx.x * blockDim.x + threadIdx.x;
    if (i < K * N) {
        int k = i / N, n = i - k * N;
        out[(size_t)n * K + k] = (__bf16)in[i];
    }
}

__global__ void zero_init_kernel(float* h0, float* s0, float* p0) {
    int i = blockIdx.x * blockDim.x + threadIdx.x;
    if (i < B_ * U_) { h0[i] = 0.0f; s0[i] = 0.0f; }
    if (i < B_ * V_) p0[i] = 0.0f;
}

// ----------------------- MX = inputs @ W_v + b_v_in ------------------------
// M = B*T = 25600 rows, K = 512, N = 768. 1600 WGs x 16 rows, 8 waves x 6 tiles.
__global__ void mx_gemm_kernel(const __bf16* __restrict__ Xbf, const __bf16* __restrict__ WvT,
                               const float* __restrict__ b_in, float* __restrict__ MX) {
    extern __shared__ char smem[];
    __bf16* As = (__bf16*)smem;                       // 16 x 512 bf16 = 16 KB
    int m0 = blockIdx.x * 16;
    int tid = threadIdx.x, lane = tid & 31, wave = tid >> 5;

    const uint4* src = (const uint4*)(Xbf + (size_t)m0 * D_);   // 16 contiguous rows
    uint4* dst = (uint4*)As;
    for (int i = tid; i < 16 * D_ / 8; i += 256) dst[i] = src[i];
    __syncthreads();

    for (int i = 0; i < 6; ++i) {
        int n0 = (wave * 6 + i) * 16;
        float bias = b_in[n0 + (lane & 15)];
        v8f acc = gemm_rowtile(As, D_, WvT, D_, n0, bias, lane);
        int n = n0 + (lane & 15);
        int mb = (lane < 16) ? 0 : 8;
#pragma unroll
        for (int r = 0; r < 8; ++r)
            MX[((size_t)(m0 + mb + r)) * TH3 + n] = acc[r];
    }
}

// ------------------------------ encoder step -------------------------------
// 16 WGs x 256 thr. LDS: MH f32 16x768 (48K) + hA bf16 16x256 (8K).
__global__ void enc_step_kernel(const float* __restrict__ h_in, float* __restrict__ h_out,
                                const float* __restrict__ MX, const __bf16* __restrict__ UvT,
                                const float* __restrict__ b_rec, float* __restrict__ out_gru_t,
                                int t) {
    extern __shared__ char smem[];
    float*  MH = (float*)smem;                        // 16 x 768
    __bf16* hA = (__bf16*)(smem + 16 * TH3 * 4);      // 16 x 256
    int b0 = blockIdx.x * 16;
    int tid = threadIdx.x, lane = tid & 31, wave = tid >> 5;

    for (int r = 0; r < 16; ++r)
        hA[r * U_ + tid] = (__bf16)h_in[(size_t)(b0 + r) * U_ + tid];
    __syncthreads();

    for (int i = 0; i < 6; ++i) {
        int n0 = (wave * 6 + i) * 16;
        float bias = b_rec[n0 + (lane & 15)];
        v8f acc = gemm_rowtile(hA, U_, UvT, U_, n0, bias, lane);
        store_c_lds(MH, TH3, n0, acc, lane);
    }
    __syncthreads();

    int col = tid;                                    // 0..255
    for (int r = 0; r < 16; ++r) {
        int b = b0 + r;
        size_t mrow = ((size_t)b * T_ + t) * TH3;
        float xz = MX[mrow + col], xr = MX[mrow + U_ + col], xh = MX[mrow + 2 * U_ + col];
        float rz = MH[r * TH3 + col], rr = MH[r * TH3 + U_ + col], rh = MH[r * TH3 + 2 * U_ + col];
        float z  = sigm(xz + rz);
        float rg = sigm(xr + rr);
        float hh = tanhf(xh + rg * rh);
        float hp = h_in[(size_t)b * U_ + col];
        float hn = z * hp + (1.0f - z) * hh;
        h_out[(size_t)b * U_ + col]     = hn;
        out_gru_t[(size_t)b * U_ + col] = hn;
    }
}

// ------------------------------ decoder GRU --------------------------------
// Fused: softmax(pred)@emb -> MX=x@W_p+b  and  MH=state@U_p+b -> gating.
// LDS: MXs 48K | MHs 48K | stateA 8K | embA 2K | P 2K  = 108 KB.
__global__ void dec_gru_kernel(const float* __restrict__ pred_in, const float* __restrict__ state_in,
                               float* __restrict__ state_out,
                               __bf16* __restrict__ state_bf, __bf16* __restrict__ state_bfT,
                               const __bf16* __restrict__ embT, const __bf16* __restrict__ WpT,
                               const __bf16* __restrict__ UpT,
                               const float* __restrict__ b_in, const float* __restrict__ b_rec) {
    extern __shared__ char smem[];
    float*  MXs    = (float*)smem;                       // 16x768
    float*  MHs    = (float*)(smem + 49152);             // 16x768
    __bf16* stateA = (__bf16*)(smem + 98304);            // 16x256
    __bf16* embA   = (__bf16*)(smem + 106496);           // 16x64
    __bf16* P      = (__bf16*)(smem + 108544);           // 16x64
    int b0 = blockIdx.x * 16;
    int tid = threadIdx.x, lane = tid & 31, wave = tid >> 5;

    for (int r = 0; r < 16; ++r)
        stateA[r * U_ + tid] = (__bf16)state_in[(size_t)(b0 + r) * U_ + tid];

    if (tid < 16) {                                   // softmax of pred rows
        const float* pr = pred_in + (size_t)(b0 + tid) * V_;
        float mx = -1e30f;
        for (int v = 0; v < V_; ++v) mx = fmaxf(mx, pr[v]);
        float s = 0.0f;
        for (int v = 0; v < V_; ++v) s += __expf(pr[v] - mx);
        float inv = 1.0f / s;
        for (int v = 0; v < V_; ++v) P[tid * V_ + v] = (__bf16)(__expf(pr[v] - mx) * inv);
    }
    __syncthreads();

    if (wave < 4) {                                   // emb_out = P @ emb  (16x64, K=64)
        int n0 = wave * 16;
        v8f acc = gemm_rowtile(P, V_, embT, V_, n0, 0.0f, lane);
        int n = n0 + (lane & 15);
        int mb = (lane < 16) ? 0 : 8;
#pragma unroll
        for (int r = 0; r < 8; ++r) embA[(mb + r) * V_ + n] = (__bf16)acc[r];
    }
    __syncthreads();

    for (int i = 0; i < 6; ++i) {
        int n0 = (wave * 6 + i) * 16;
        int nl = n0 + (lane & 15);
        v8f accx = gemm_rowtile(embA, V_, WpT, V_, n0, b_in[nl], lane);
        store_c_lds(MXs, TH3, n0, accx, lane);
        v8f acch = gemm_rowtile(stateA, U_, UpT, U_, n0, b_rec[nl], lane);
        store_c_lds(MHs, TH3, n0, acch, lane);
    }
    __syncthreads();

    int col = tid;
    for (int r = 0; r < 16; ++r) {
        int b = b0 + r;
        float xz = MXs[r * TH3 + col], xr = MXs[r * TH3 + U_ + col], xh = MXs[r * TH3 + 2 * U_ + col];
        float rz = MHs[r * TH3 + col], rr = MHs[r * TH3 + U_ + col], rh = MHs[r * TH3 + 2 * U_ + col];
        float z  = sigm(xz + rz);
        float rg = sigm(xr + rr);
        float hh = tanhf(xh + rg * rh);
        float sp = state_in[(size_t)b * U_ + col];
        float sn = z * sp + (1.0f - z) * hh;
        state_out[(size_t)b * U_ + col] = sn;
        state_bf[(size_t)b * U_ + col]  = (__bf16)sn;
        state_bfT[(size_t)col * B_ + b] = (__bf16)sn;     // transposed mirror for attn GEMM
    }
}

// --------------------------- decoder attention -----------------------------
// scores = q@state^T ; P = softmax ; attn = P@state ; context = state*attn ;
// pred = softmax([state|context]@W3 + b3) -> pred buffer + d_out[b][t][v].
// LDS: S 16K | AT 16K | LG 4K | qA 8K | Pb 8K | CC 16K = 68 KB.
__global__ void dec_attn_kernel(const float* __restrict__ q, const float* __restrict__ state,
                                const __bf16* __restrict__ state_bf, const __bf16* __restrict__ state_bfT,
                                const __bf16* __restrict__ W3T, const float* __restrict__ b3,
                                float* __restrict__ pred_out, float* __restrict__ out, int t) {
    extern __shared__ char smem[];
    float*  S  = (float*)smem;                         // 16x256 scores
    float*  AT = (float*)(smem + 16384);               // 16x256 attn
    float*  LG = (float*)(smem + 32768);               // 16x64 logits
    __bf16* qA = (__bf16*)(smem + 36864);              // 16x256
    __bf16* Pb = (__bf16*)(smem + 45056);              // 16x256
    __bf16* CC = (__bf16*)(smem + 53248);              // 16x512
    int b0 = blockIdx.x * 16;
    int tid = threadIdx.x, lane = tid & 31, wave = tid >> 5;

    for (int r = 0; r < 16; ++r)
        qA[r * U_ + tid] = (__bf16)q[(size_t)(b0 + r) * U_ + tid];
    __syncthreads();

    for (int i = 0; i < 2; ++i) {                     // scores: N=256 (batch j), K=256 (u)
        int n0 = (wave * 2 + i) * 16;
        v8f acc = gemm_rowtile(qA, U_, state_bf, U_, n0, 0.0f, lane);
        store_c_lds(S, B_, n0, acc, lane);
    }
    __syncthreads();

    if (tid < 16) {                                   // row softmax over 256
        float mx = -1e30f;
        for (int j = 0; j < B_; ++j) mx = fmaxf(mx, S[tid * B_ + j]);
        float s = 0.0f;
        for (int j = 0; j < B_; ++j) { float e = __expf(S[tid * B_ + j] - mx); S[tid * B_ + j] = e; s += e; }
        float inv = 1.0f / s;
        for (int j = 0; j < B_; ++j) Pb[tid * B_ + j] = (__bf16)(S[tid * B_ + j] * inv);
    }
    __syncthreads();

    for (int i = 0; i < 2; ++i) {                     // attn: N=256 (u), K=256 (j)
        int n0 = (wave * 2 + i) * 16;
        v8f acc = gemm_rowtile(Pb, B_, state_bfT, B_, n0, 0.0f, lane);
        store_c_lds(AT, U_, n0, acc, lane);
    }
    __syncthreads();

    {
        int col = tid;                                // concat [state | state*attn] as bf16
        for (int r = 0; r < 16; ++r) {
            float sv = state[(size_t)(b0 + r) * U_ + col];
            float av = AT[r * U_ + col];
            CC[r * 512 + col]       = (__bf16)sv;
            CC[r * 512 + U_ + col]  = (__bf16)(sv * av);
        }
    }
    __syncthreads();

    if (wave < 4) {                                   // logits: N=64, K=512
        int n0 = wave * 16;
        v8f acc = gemm_rowtile(CC, 512, W3T, 512, n0, b3[n0 + (lane & 15)], lane);
        store_c_lds(LG, V_, n0, acc, lane);
    }
    __syncthreads();

    if (tid < 16) {                                   // final row softmax over V=64
        float mx = -1e30f;
        for (int v = 0; v < V_; ++v) mx = fmaxf(mx, LG[tid * V_ + v]);
        float s = 0.0f;
        for (int v = 0; v < V_; ++v) s += __expf(LG[tid * V_ + v] - mx);
        float inv = 1.0f / s;
        int b = b0 + tid;
        for (int v = 0; v < V_; ++v) {
            float p = __expf(LG[tid * V_ + v] - mx) * inv;
            pred_out[(size_t)b * V_ + v]        = p;
            out[((size_t)b * T_ + t) * V_ + v]  = p;   // d_out is [B,T,V]
        }
    }
}

// ------------------------------- launcher ----------------------------------

extern "C" void kernel_launch(void* const* d_in, const int* in_sizes, int n_in,
                              void* d_out, int out_size, void* d_ws, size_t ws_size,
                              hipStream_t stream) {
    const float* inputs  = (const float*)d_in[0];   // [B,T,D]
    const float* W_v     = (const float*)d_in[1];   // [D,768]
    const float* U_v     = (const float*)d_in[2];   // [U,768]
    const float* b_v_in  = (const float*)d_in[3];
    const float* b_v_rec = (const float*)d_in[4];
    const float* W_p     = (const float*)d_in[5];   // [V,768]
    const float* U_p     = (const float*)d_in[6];   // [U,768]
    const float* b_p_in  = (const float*)d_in[7];
    const float* b_p_rec = (const float*)d_in[8];
    const float* emb     = (const float*)d_in[9];   // [V,V]
    const float* W3      = (const float*)d_in[10];  // [2U,V]
    const float* b3      = (const float*)d_in[11];

    // workspace carve (256B aligned)
    char* w = (char*)d_ws;
    auto carve = [&](size_t bytes) { char* p = w; w += (bytes + 255) & ~(size_t)255; return p; };
    __bf16* Xbf   = (__bf16*)carve((size_t)B_ * T_ * D_ * 2);
    __bf16* WvT   = (__bf16*)carve((size_t)TH3 * D_ * 2);
    __bf16* UvT   = (__bf16*)carve((size_t)TH3 * U_ * 2);
    __bf16* WpT   = (__bf16*)carve((size_t)TH3 * V_ * 2);
    __bf16* UpT   = (__bf16*)carve((size_t)TH3 * U_ * 2);
    __bf16* embT  = (__bf16*)carve((size_t)V_ * V_ * 2);
    __bf16* W3T   = (__bf16*)carve((size_t)V_ * 2 * U_ * 2);
    float*  MX    = (float*)carve((size_t)B_ * T_ * TH3 * 4);
    float*  outg  = (float*)carve((size_t)T_ * B_ * U_ * 4);
    float*  hbuf  = (float*)carve((size_t)2 * B_ * U_ * 4);
    float*  sbuf  = (float*)carve((size_t)2 * B_ * U_ * 4);
    float*  pbuf  = (float*)carve((size_t)2 * B_ * V_ * 4);
    __bf16* sbf   = (__bf16*)carve((size_t)B_ * U_ * 2);
    __bf16* sbfT  = (__bf16*)carve((size_t)U_ * B_ * 2);

    (void)in_sizes; (void)n_in; (void)out_size; (void)ws_size;

    // raise dynamic-LDS caps for the big fused kernels (>64KB)
    hipFuncSetAttribute((const void*)dec_gru_kernel,
                        hipFuncAttributeMaxDynamicSharedMemorySize, 110592);
    hipFuncSetAttribute((const void*)dec_attn_kernel,
                        hipFuncAttributeMaxDynamicSharedMemorySize, 69632);

    // 1) precision conversion (weights transposed to [N][K] bf16)
    int nX = B_ * T_ * D_;
    cvt_bf16_kernel<<<(nX + 255) / 256, 256, 0, stream>>>(inputs, Xbf, nX);
    cvtT_bf16_kernel<<<(D_ * TH3 + 255) / 256, 256, 0, stream>>>(W_v, WvT, D_, TH3);
    cvtT_bf16_kernel<<<(U_ * TH3 + 255) / 256, 256, 0, stream>>>(U_v, UvT, U_, TH3);
    cvtT_bf16_kernel<<<(V_ * TH3 + 255) / 256, 256, 0, stream>>>(W_p, WpT, V_, TH3);
    cvtT_bf16_kernel<<<(U_ * TH3 + 255) / 256, 256, 0, stream>>>(U_p, UpT, U_, TH3);
    cvtT_bf16_kernel<<<(V_ * V_ + 255) / 256, 256, 0, stream>>>(emb, embT, V_, V_);
    cvtT_bf16_kernel<<<(2 * U_ * V_ + 255) / 256, 256, 0, stream>>>(W3, W3T, 2 * U_, V_);

    // 2) zero initial states (every call: ws is not re-poisoned between replays)
    zero_init_kernel<<<(B_ * U_ + 255) / 256, 256, 0, stream>>>(hbuf, sbuf, pbuf);

    // 3) big parallel GEMM: MX[b*T+t] = x_{b,t} @ W_v + b_v_in
    mx_gemm_kernel<<<(B_ * T_) / 16, 256, 16 * D_ * 2, stream>>>(Xbf, WvT, b_v_in, MX);

    // 4) encoder scan (ping-pong h)
    for (int t = 0; t < T_; ++t) {
        enc_step_kernel<<<16, 256, 16 * TH3 * 4 + 16 * U_ * 2, stream>>>(
            hbuf + (size_t)(t & 1) * B_ * U_,
            hbuf + (size_t)((t + 1) & 1) * B_ * U_,
            MX, UvT, b_v_rec,
            outg + (size_t)t * B_ * U_, t);
    }

    // 5) decoder scan (ping-pong state/pred)
    for (int t = 0; t < T_; ++t) {
        const float* st_in  = sbuf + (size_t)(t & 1) * B_ * U_;
        float*       st_out = sbuf + (size_t)((t + 1) & 1) * B_ * U_;
        const float* pd_in  = pbuf + (size_t)(t & 1) * B_ * V_;
        float*       pd_out = pbuf + (size_t)((t + 1) & 1) * B_ * V_;

        dec_gru_kernel<<<16, 256, 110592, stream>>>(
            pd_in, st_in, st_out, sbf, sbfT, embT, WpT, UpT, b_p_in, b_p_rec);

        dec_attn_kernel<<<16, 256, 69632, stream>>>(
            outg + (size_t)t * B_ * U_, st_out, sbf, sbfT, W3T, b3,
            pd_out, (float*)d_out, t);
    }
}
